// TopKRouter_57921928954061
// MI455X (gfx1250) — matrix operand order repaired and checked by reading, not compiled, not run on hardware
//
#include <hip/hip_runtime.h>
#include <math.h>

typedef __attribute__((ext_vector_type(16))) __bf16 v16bf;
typedef __attribute__((ext_vector_type(8)))  float  v8f;
typedef __attribute__((ext_vector_type(4)))  float  f32x4;

#define TOK_TILE 64
#define IN_DIM   1024
#define HID      256
#define NEXP     16
#define SX_STRIDE (IN_DIM + 8)              // 1032 bf16: row stride 2064 B -> uniform (optimal) 2-phase b128 LDS access
#define SH_STRIDE (HID + 8)                 // 264 bf16: row stride 528 B, 16B aligned
#define SX_BYTES  (TOK_TILE * SX_STRIDE * 2)   // 132096 B -> 2 blocks per WGP (320KB LDS)
#define SH_BYTES  (TOK_TILE * SH_STRIDE * 2)   // 33792 B (aliased over sx after GEMM1)
#define SMEM_BYTES SX_BYTES

__device__ __forceinline__ unsigned short f32_to_bf16(float f) {
    union { float f; unsigned u; } c; c.f = f;
    unsigned u = c.u;
    u += 0x7FFFu + ((u >> 16) & 1u);        // round-to-nearest-even
    return (unsigned short)(u >> 16);
}

union FragU {
    uint4 q[2];
    v16bf v;
};

#define WMMA_BF16(A, B, C) \
    __builtin_amdgcn_wmma_f32_16x16x32_bf16(false, (A).v, false, (B).v, (short)0, (C), false, false)

// ---- prep: W1 (1024x256 f32 row-major) -> w1t (bf16, [n][k]); W2 -> w2t (bf16, [n][k]) ----
__global__ __launch_bounds__(512) void router_prep(const float* __restrict__ W1,
                                                   const float* __restrict__ W2,
                                                   unsigned short* __restrict__ w1t,
                                                   unsigned short* __restrict__ w2t) {
    int idx = blockIdx.x * 512 + threadIdx.x;
    if (idx < IN_DIM * HID) {
        int k = idx / HID, n = idx % HID;
        w1t[(size_t)n * IN_DIM + k] = f32_to_bf16(W1[idx]);
    } else if (idx - IN_DIM * HID < HID * NEXP) {
        int j = idx - IN_DIM * HID;
        int k = j / NEXP, n = j % NEXP;
        w2t[(size_t)n * HID + k] = f32_to_bf16(W2[j]);
    }
}

// ---- main: 64 tokens per block, 16 waves as 8(col-pairs) x 2(row-pairs).
// Each wave: 2 col tiles x 2 row tiles -> 4 WMMAs per K-step from 2 A-frags (LDS) + 2 B-frags (L2).
__global__ __launch_bounds__(512) void router_main(const float* __restrict__ x,
                                                   const float* __restrict__ b1,
                                                   const float* __restrict__ b2,
                                                   const unsigned short* __restrict__ w1t,
                                                   const unsigned short* __restrict__ w2t,
                                                   float* __restrict__ alpha_out,
                                                   float* __restrict__ logits_out) {
    extern __shared__ unsigned char smem[];
    unsigned short* sx  = (unsigned short*)smem;             // 64 x 1032 bf16 (phase 1)
    unsigned short* sh  = (unsigned short*)smem;             // 64 x 264 bf16  (phase 2, aliases sx)
    float*          slog = (float*)(smem + SH_BYTES);        // 64 x 16 f32    (phase 2)

    const int t    = threadIdx.x;
    const int lane = t & 31;
    const int wave = t >> 5;
    const int g    = lane >> 4;      // half-wave group (ISA fragment layout)
    const int mrow = lane & 15;      // A row / B,C,D column index within a 16x16 tile
    const long row0 = (long)blockIdx.x * TOK_TILE;

    // ---- stage x tile (64 x 1024 f32) -> LDS bf16; non-temporal (x has zero reuse) ----
    {
        const f32x4* src = (const f32x4*)(x + row0 * IN_DIM);   // 16384 float4, fully coalesced
        #pragma unroll 4
        for (int j = 0; j < 32; ++j) {
            int i  = j * 512 + t;
            int r  = i >> 8;          // row 0..63
            int c4 = i & 255;         // float4 index in row
            f32x4 v = __builtin_nontemporal_load(src + i);
            uint2 packed;
            packed.x = (unsigned)f32_to_bf16(v.x) | ((unsigned)f32_to_bf16(v.y) << 16);
            packed.y = (unsigned)f32_to_bf16(v.z) | ((unsigned)f32_to_bf16(v.w) << 16);
            *(uint2*)(&sx[r * SX_STRIDE + c4 * 4]) = packed;
        }
    }
    __syncthreads();

    // ---- GEMM1: 4 accumulators per wave (2 row tiles x 2 col tiles), K = 1024 ----
    const int c0   = (wave & 7) * 2;          // first of 2 column tiles
    const int r0   = (wave >> 3) * 2;         // first of 2 row tiles
    const int colA = c0 * 16 + mrow;
    const int colB = colA + 16;
    v8f acc00 = {}, acc01 = {}, acc10 = {}, acc11 = {};
    {
        const unsigned short* arow0 = sx + (r0 * 16 + mrow) * SX_STRIDE;
        const unsigned short* arow1 = arow0 + 16 * SX_STRIDE;
        const unsigned short* bcol0 = w1t + (size_t)colA * IN_DIM + 16 * g;
        const unsigned short* bcol1 = w1t + (size_t)colB * IN_DIM + 16 * g;
        for (int kb = 0; kb < IN_DIM; kb += 32) {
            FragU A0, A1, B0, B1;
            A0.q[0] = *(const uint4*)(arow0 + kb + 16 * g);
            A0.q[1] = *(const uint4*)(arow0 + kb + 16 * g + 8);
            A1.q[0] = *(const uint4*)(arow1 + kb + 16 * g);
            A1.q[1] = *(const uint4*)(arow1 + kb + 16 * g + 8);
            const uint4* p0 = (const uint4*)(bcol0 + kb);
            B0.q[0] = p0[0]; B0.q[1] = p0[1];
            const uint4* p1 = (const uint4*)(bcol1 + kb);
            B1.q[0] = p1[0]; B1.q[1] = p1[1];
            acc00 = WMMA_BF16(A0, B0, acc00);
            acc01 = WMMA_BF16(A0, B1, acc01);
            acc10 = WMMA_BF16(A1, B0, acc10);
            acc11 = WMMA_BF16(A1, B1, acc11);
        }
    }
    __syncthreads();   // everyone done reading sx; safe to alias it with sh

    // ---- bias + ELU, write h tiles (bf16) into aliased LDS ----
    {
        float bias0 = b1[colA];
        float bias1 = b1[colB];
        #pragma unroll
        for (int j = 0; j < 8; ++j) {
            int m0 = r0 * 16 + j + 8 * g;     // row in tile r0
            int m1 = m0 + 16;                 // row in tile r0+1
            float v00 = acc00[j] + bias0, v01 = acc01[j] + bias1;
            float v10 = acc10[j] + bias0, v11 = acc11[j] + bias1;
            sh[m0 * SH_STRIDE + colA] = f32_to_bf16(v00 > 0.f ? v00 : __expf(v00) - 1.f);
            sh[m0 * SH_STRIDE + colB] = f32_to_bf16(v01 > 0.f ? v01 : __expf(v01) - 1.f);
            sh[m1 * SH_STRIDE + colA] = f32_to_bf16(v10 > 0.f ? v10 : __expf(v10) - 1.f);
            sh[m1 * SH_STRIDE + colB] = f32_to_bf16(v11 > 0.f ? v11 : __expf(v11) - 1.f);
        }
    }
    __syncthreads();

    // ---- GEMM2 (waves 0..3): logits[16w:16w+16, :] = h @ W2 + b2 ----
    if (wave < 4) {
        v8f acc2 = {};
        const unsigned short* arow = sh + (wave * 16 + mrow) * SH_STRIDE;
        const unsigned short* bcol = w2t + (size_t)mrow * HID + 16 * g;
        #pragma unroll
        for (int kb = 0; kb < HID; kb += 32) {
            FragU A, B;
            A.q[0] = *(const uint4*)(arow + kb + 16 * g);
            A.q[1] = *(const uint4*)(arow + kb + 16 * g + 8);
            const uint4* pb = (const uint4*)(bcol + kb);
            B.q[0] = pb[0]; B.q[1] = pb[1];
            acc2 = WMMA_BF16(A, B, acc2);
        }
        float bias2 = b2[mrow];
        #pragma unroll
        for (int j = 0; j < 8; ++j) {
            float lg = acc2[j] + bias2;
            int m = wave * 16 + j + 8 * g;
            slog[m * NEXP + mrow] = lg;
            __builtin_nontemporal_store(lg, &logits_out[(row0 + m) * NEXP + mrow]);
        }
    }
    __syncthreads();

    // ---- top-2 masked softmax, one lane per token row ----
    if (t < TOK_TILE) {
        float best = -INFINITY, second = -INFINITY;
        int bi = -1, si = -1;
        #pragma unroll
        for (int e = 0; e < NEXP; ++e) {
            float lv = slog[t * NEXP + e];
            if (lv > best)        { second = best; si = bi; best = lv; bi = e; }
            else if (lv > second) { second = lv; si = e; }
        }
        float d  = __expf(second - best);
        float z  = 1.0f + d;
        float a1 = 1.0f / z;
        float a2 = d / z;
        float* dst = alpha_out + (row0 + t) * NEXP;
        #pragma unroll
        for (int e = 0; e < NEXP; ++e) {
            float a = (e == bi) ? a1 : ((e == si) ? a2 : 0.0f);
            __builtin_nontemporal_store(a, &dst[e]);
        }
    }
}

extern "C" void kernel_launch(void* const* d_in, const int* in_sizes, int n_in,
                              void* d_out, int out_size, void* d_ws, size_t ws_size,
                              hipStream_t stream) {
    const float* x  = (const float*)d_in[0];
    const float* W1 = (const float*)d_in[1];
    const float* b1 = (const float*)d_in[2];
    const float* W2 = (const float*)d_in[3];
    const float* b2 = (const float*)d_in[4];

    const int Ntok = in_sizes[0] / IN_DIM;       // 262144

    unsigned short* w1t = (unsigned short*)d_ws;              // 512 KB
    unsigned short* w2t = w1t + (size_t)IN_DIM * HID;         // 8 KB
    float* alpha_out  = (float*)d_out;
    float* logits_out = alpha_out + (size_t)Ntok * NEXP;

    int prep_total = IN_DIM * HID + HID * NEXP;
    router_prep<<<(prep_total + 511) / 512, 512, 0, stream>>>(W1, W2, w1t, w2t);
    router_main<<<Ntok / TOK_TILE, 512, SMEM_BYTES, stream>>>(x, b1, b2, w1t, w2t,
                                                              alpha_out, logits_out);
}
